// UnifiedInterlacer_7876970021341
// MI455X (gfx1250) — matrix operand order repaired and checked by reading, not compiled, not run on hardware
//
#include <hip/hip_runtime.h>

// ---------------- problem constants (from reference) ----------------
#define BB   8
#define NN   8192
#define KNN  6
#define DIN  3
#define DD   256
#define LL   5
#define MM   (BB * NN)   // 65536 rows
#define D3   (3 * DD)    // 768

// ---------------- CDNA5 WMMA types ----------------
typedef __attribute__((ext_vector_type(16))) __bf16 v16bf;
typedef __attribute__((ext_vector_type(8)))  __bf16 v8bf;
typedef __attribute__((ext_vector_type(8)))  float  v8f;

// ---------------- async global->LDS path (gfx1250), guarded ----------------
#if defined(__has_builtin)
#if __has_builtin(__builtin_amdgcn_global_load_async_to_lds_b128) && \
    __has_builtin(__builtin_amdgcn_s_wait_asynccnt)
#define HAVE_ASYNC_LDS 1
#endif
#endif
#ifndef HAVE_ASYNC_LDS
#define HAVE_ASYNC_LDS 0
#endif

#if HAVE_ASYNC_LDS
// builtin signature (from clang diagnostic): (v4i as1*, v4i as3*, imm, imm)
typedef int v4i __attribute__((vector_size(16)));
#define ASG __attribute__((address_space(1)))
#define ASL __attribute__((address_space(3)))
static __device__ __forceinline__ ASG v4i* gcast4(const void* p) {
  return (ASG v4i*)(unsigned long long)(size_t)p;
}
// generic LDS pointer low 32 bits == LDS offset (shared aperture mapping)
static __device__ __forceinline__ ASL v4i* lcast4(const void* p) {
  return (ASL v4i*)(unsigned int)(size_t)p;
}
#endif

// epilogue flags
enum {
  EF_BIAS       = 1,
  EF_RELU       = 2,   // max(x,0)+1e-6 (linear-attention feature map)
  EF_ROWSCALE   = 4,   // multiply row m by rowscale[b*strideRS + m]
  EF_RESID      = 8,   // resid[idx] += acc (residual update of h)
  EF_WRITE_F32  = 16,
  EF_WRITE_BF16 = 32,
  EF_TRANS_BF16 = 64,  // store bf16 transposed per batch: [B, Nc, NperB]
};

#define WTP 72   // padded LDS row stride (bf16) for the 32x64 W tile

// ---------------------------------------------------------------------------
// Generic batched GEMM:  C[b] = A[b] (M x K, bf16 row-major)
//                               @ W[b][:, colOff:colOff+64*gx] (K x ldb, bf16)
// 256 threads = 8 waves; block computes a 128(M) x 64(N) tile.
// W k-slab (32x64) is staged once per block into LDS (async, double-buffered);
// each wave runs 4x v_wmma_f32_16x16x32_bf16 per K-step. Fragment loads are
// split from WMMA issue so the ds_loads clause up and drain with one wait.
// ---------------------------------------------------------------------------
__global__ __launch_bounds__(256) void wmma_gemm_bf16(
    const __bf16* __restrict__ A, int lda, long long strideA,
    const __bf16* __restrict__ W, int ldb, int colOff, long long strideB,
    const float* __restrict__ bias,
    const float* __restrict__ rowscale, long long strideRS,
    float* __restrict__ Cf, __bf16* __restrict__ Cbf, float* __restrict__ resid,
    int ldc, long long strideC, int M, int Kdim, int NperB, int flags)
{
  __shared__ __bf16 wt[2][32 * WTP];

  const int lane = threadIdx.x & 31;
  const int wid  = threadIdx.x >> 5;           // 0..7
  const int b    = blockIdx.z;
  const int tm   = blockIdx.y * 128 + wid * 16;
  const int tn   = blockIdx.x * 64;

  A += (long long)b * strideA;
  W += (long long)b * strideB;

  // ---- W-tile staging coordinates: thread -> (row 0..31, 16B segment 0..7)
  const int srow = threadIdx.x >> 3;
  const int sseg = threadIdx.x & 7;
  const __bf16* gsrc = W + (long long)srow * ldb + colOff + tn + sseg * 8;
  __bf16* ldst[2] = { &wt[0][srow * WTP + sseg * 8],
                      &wt[1][srow * WTP + sseg * 8] };

  auto stage = [&](int buf, int k0) {
    const __bf16* src = gsrc + (long long)k0 * ldb;
#if HAVE_ASYNC_LDS
    __builtin_amdgcn_global_load_async_to_lds_b128(gcast4(src),
                                                   lcast4(ldst[buf]), 0, 0);
#else
    *(v8bf*)ldst[buf] = *(const v8bf*)src;    // fallback: VGPR bounce
#endif
  };

  // ---- A fragment addressing (16-bit A 16x32 layout):
  //   lanes 0-15 : rows m=tm+lane,    K chunks [0..7]  and [16..23]
  //   lanes16-31 : rows m=tm+lane-16, K chunks [8..15] and [24..31]
  const int am  = tm + (lane & 15);
  const int klo = (lane >> 4) * 8;
  const __bf16* arow = A + (long long)am * lda;

  v8f acc[4] = {};

  stage(0, 0);
  const int nsteps = Kdim >> 5;
  for (int s = 0; s < nsteps; ++s) {
    const int k0  = s << 5;
    const int buf = s & 1;
#if HAVE_ASYNC_LDS
    __builtin_amdgcn_s_wait_asynccnt(0);
#endif
    __syncthreads();                 // tile[buf] visible; tile[buf^1] free
    if (s + 1 < nsteps) stage(buf ^ 1, k0 + 32);

    // ---- load phase: A fragment (global) + 4 B fragments (LDS) ----
    v8bf alo = *(const v8bf*)(arow + k0 + klo);
    v8bf ahi = *(const v8bf*)(arow + k0 + 16 + klo);
    v16bf af = __builtin_shufflevector(alo, ahi,
                                       0, 1, 2, 3, 4, 5, 6, 7,
                                       8, 9, 10, 11, 12, 13, 14, 15);
    if (s + 1 < nsteps) __builtin_prefetch(arow + k0 + 32, 0, 1);

    const __bf16* wr = &wt[buf][((lane >> 4) * 8) * WTP + (lane & 15)];
    v16bf bf0, bf1, bf2, bf3;
    {
#pragma unroll
      for (int e = 0; e < 8; ++e) {
        bf0[e]     = wr[e * WTP + 0];
        bf0[8 + e] = wr[(16 + e) * WTP + 0];
        bf1[e]     = wr[e * WTP + 16];
        bf1[8 + e] = wr[(16 + e) * WTP + 16];
        bf2[e]     = wr[e * WTP + 32];
        bf2[8 + e] = wr[(16 + e) * WTP + 32];
        bf3[e]     = wr[e * WTP + 48];
        bf3[8 + e] = wr[(16 + e) * WTP + 48];
      }
    }

    // ---- compute phase: 4 back-to-back WMMAs ----
    acc[0] = __builtin_amdgcn_wmma_f32_16x16x32_bf16(false, af, false, bf0,
                                                     (short)0, acc[0],
                                                     false, false);
    acc[1] = __builtin_amdgcn_wmma_f32_16x16x32_bf16(false, af, false, bf1,
                                                     (short)0, acc[1],
                                                     false, false);
    acc[2] = __builtin_amdgcn_wmma_f32_16x16x32_bf16(false, af, false, bf2,
                                                     (short)0, acc[2],
                                                     false, false);
    acc[3] = __builtin_amdgcn_wmma_f32_16x16x32_bf16(false, af, false, bf3,
                                                     (short)0, acc[3],
                                                     false, false);
  }

  // Epilogue. C/D layout: VGPR r -> row tm + r + 8*(lane>=16), col tn+16j+(lane&15)
  const int nlo = lane & 15;
  const int mhi = (lane >> 4) * 8;
#pragma unroll
  for (int j = 0; j < 4; ++j) {
    const int n = tn + j * 16 + nlo;
#pragma unroll
    for (int r = 0; r < 8; ++r) {
      const int m = tm + mhi + r;
      float v = acc[j][r];
      if (flags & EF_ROWSCALE) v *= rowscale[(long long)b * strideRS + m];
      if (flags & EF_BIAS)     v += bias[n];
      if (flags & EF_RELU)     v = fmaxf(v, 0.0f) + 1e-6f;
      const long long cidx = (long long)b * strideC + (long long)m * ldc + n;
      if (flags & EF_RESID) {
        float h2 = resid[cidx] + v;
        resid[cidx] = h2;
        v = h2;
      }
      if (flags & EF_WRITE_F32)  Cf[cidx]  = v;
      if (flags & EF_WRITE_BF16) Cbf[cidx] = (__bf16)v;
      if (flags & EF_TRANS_BF16) {
        const int bb = m / NperB;
        const int nn = m - bb * NperB;
        Cbf[(long long)bb * ldc * NperB + (long long)n * NperB + nn] = (__bf16)v;
      }
    }
  }
}

// ---------------- LayerNorm over D=256, writes bf16 ----------------
__global__ __launch_bounds__(256) void ln_bf16_kernel(
    const float* __restrict__ h, const float* __restrict__ g,
    const float* __restrict__ be, __bf16* __restrict__ out)
{
  __shared__ float red[256];
  const long long row = blockIdx.x;
  const int d = threadIdx.x;
  const float x = h[row * DD + d];
  red[d] = x;
  __syncthreads();
  for (int s = 128; s > 0; s >>= 1) { if (d < s) red[d] += red[d + s]; __syncthreads(); }
  const float mean = red[0] * (1.0f / DD);
  __syncthreads();
  const float c = x - mean;
  red[d] = c * c;
  __syncthreads();
  for (int s = 128; s > 0; s >>= 1) { if (d < s) red[d] += red[d + s]; __syncthreads(); }
  const float var = red[0] * (1.0f / DD);
  out[row * DD + d] = (__bf16)(c * rsqrtf(var + 1e-5f) * g[d] + be[d]);
}

// ---------------- spmm_mean scatter hop: dst[knn[n][k]] += src[n]/Kdiv --------
__global__ __launch_bounds__(256) void scatter_mean_kernel(
    const float* __restrict__ src, const int* __restrict__ knn,
    float* __restrict__ dst, float invk)
{
  const long long row = blockIdx.x;            // b*N + n
  const int d = threadIdx.x;
  const long long b = row >> 13;               // N = 8192
  const float val = src[row * DD + d] * invk;
  const int* nb = knn + row * KNN;
#pragma unroll
  for (int kk = 0; kk < KNN; ++kk) {
    const long long j = nb[kk];
    atomicAdd(&dst[((b << 13) + j) * DD + d], val);
  }
}

// ---------------- GRF: u = sum_d(q*k) * v, bf16 out ----------------
__global__ __launch_bounds__(256) void grf_attnmul_kernel(
    const float* __restrict__ q, const float* __restrict__ k,
    const float* __restrict__ v, __bf16* __restrict__ u)
{
  __shared__ float red[256];
  const long long row = blockIdx.x;
  const int d = threadIdx.x;
  red[d] = q[row * DD + d] * k[row * DD + d];
  __syncthreads();
  for (int s = 128; s > 0; s >>= 1) { if (d < s) red[d] += red[d + s]; __syncthreads(); }
  const float dot = red[0];
  u[row * DD + d] = (__bf16)(dot * v[row * DD + d]);
}

// ---------------- ksum[b,d] = sum_n kT[b,d,n] ----------------
__global__ __launch_bounds__(256) void ksum_kernel(
    const __bf16* __restrict__ kT, float* __restrict__ ksum)
{
  __shared__ float red[256];
  const long long idx = blockIdx.x;            // b*D + d
  const __bf16* p = kT + idx * NN;
  float s = 0.0f;
  for (int n = threadIdx.x; n < NN; n += 256) s += (float)p[n];
  red[threadIdx.x] = s;
  __syncthreads();
  for (int t = 128; t > 0; t >>= 1) { if (threadIdx.x < t) red[threadIdx.x] += red[threadIdx.x + t]; __syncthreads(); }
  if (threadIdx.x == 0) ksum[idx] = red[0];
}

// ---------------- z[b,n] = 1/(q . ksum + 1e-6) ----------------
__global__ __launch_bounds__(256) void z_kernel(
    const __bf16* __restrict__ q, const float* __restrict__ ksum,
    float* __restrict__ z)
{
  __shared__ float red[256];
  const long long row = blockIdx.x;
  const long long b = row >> 13;
  const int d = threadIdx.x;
  red[d] = (float)q[row * DD + d] * ksum[b * DD + d];
  __syncthreads();
  for (int s = 128; s > 0; s >>= 1) { if (d < s) red[d] += red[d + s]; __syncthreads(); }
  if (d == 0) z[row] = 1.0f / (red[0] + 1e-6f);
}

// ---------------- embed: h = x @ emb_w + emb_b (K=3, scalar FMA) ----------------
__global__ __launch_bounds__(256) void embed_kernel(
    const float* __restrict__ x, const float* __restrict__ ew,
    const float* __restrict__ eb, float* __restrict__ h)
{
  const long long i = (long long)blockIdx.x * 256 + threadIdx.x;   // over M*D
  const long long row = i >> 8;
  const int d = i & 255;
  h[i] = x[row * DIN + 0] * ew[0 * DD + d] +
         x[row * DIN + 1] * ew[1 * DD + d] +
         x[row * DIN + 2] * ew[2 * DD + d] + eb[d];
}

// ---------------- head: out = h @ head_w[256,3] + head_b ----------------
__global__ __launch_bounds__(256) void head_kernel(
    const float* __restrict__ h, const float* __restrict__ w,
    const float* __restrict__ bias, float* __restrict__ out)
{
  __shared__ float red[256];
  const long long row = blockIdx.x;
  const int d = threadIdx.x;
  const float x = h[row * DD + d];
#pragma unroll
  for (int j = 0; j < 3; ++j) {
    red[d] = x * w[d * 3 + j];
    __syncthreads();
    for (int s = 128; s > 0; s >>= 1) { if (d < s) red[d] += red[d + s]; __syncthreads(); }
    if (d == 0) out[row * 3 + j] = red[0] + bias[j];
    __syncthreads();
  }
}

// ---------------- fp32 -> bf16 convert ----------------
__global__ __launch_bounds__(256) void f2bf_kernel(
    const float* __restrict__ src, __bf16* __restrict__ dst, long long n)
{
  const long long i = (long long)blockIdx.x * 256 + threadIdx.x;
  if (i < n) dst[i] = (__bf16)src[i];
}

// ---------------------------------------------------------------------------
// Host orchestration
// ---------------------------------------------------------------------------
static inline void gemm(hipStream_t s,
                        const __bf16* A, int lda, long long sA,
                        const __bf16* W, int ldb, int colOff, long long sB,
                        const float* bias, const float* rowscale, long long sRS,
                        float* Cf, __bf16* Cbf, float* resid,
                        int ldc, long long sC,
                        int M, int Kdim, int Nc, int nbatch, int NperB, int flags)
{
  dim3 g(Nc / 64, M / 128, nbatch), blk(256, 1, 1);
  wmma_gemm_bf16<<<g, blk, 0, s>>>(A, lda, sA, W, ldb, colOff, sB, bias,
                                   rowscale, sRS, Cf, Cbf, resid, ldc, sC,
                                   M, Kdim, NperB, flags);
}

extern "C" void kernel_launch(void* const* d_in, const int* in_sizes, int n_in,
                              void* d_out, int out_size, void* d_ws, size_t ws_size,
                              hipStream_t stream)
{
  (void)in_sizes; (void)n_in; (void)out_size; (void)ws_size;

  const float* x        = (const float*)d_in[0];
  const int*   knn      = (const int*)  d_in[1];
  const float* emb_w    = (const float*)d_in[2];
  const float* emb_b    = (const float*)d_in[3];
  const float* norm_g   = (const float*)d_in[4];
  const float* norm_b   = (const float*)d_in[5];
  const float* grf_qkv  = (const float*)d_in[6];
  const float* grf_outw = (const float*)d_in[7];
  const float* grf_outb = (const float*)d_in[8];
  const float* attn_qkv = (const float*)d_in[9];
  const float* attn_outw= (const float*)d_in[10];
  const float* attn_outb= (const float*)d_in[11];
  const float* head_w   = (const float*)d_in[12];
  const float* head_b   = (const float*)d_in[13];
  float* out = (float*)d_out;

  // workspace carve-up (256B aligned)
  char* p = (char*)d_ws;
  auto take = [&](size_t bytes) -> char* {
    char* r = p;
    p += (bytes + 255) & ~(size_t)255;
    return r;
  };
  float*  h    = (float*)take((size_t)MM * DD * 4);   // running hidden state
  float*  qf   = (float*)take((size_t)MM * DD * 4);   // GRF q (fp32)
  float*  t0   = (float*)take((size_t)MM * DD * 4);   // scatter ping
  float*  t1   = (float*)take((size_t)MM * DD * 4);   // scatter pong (k final)
  float*  t2   = (float*)take((size_t)MM * DD * 4);   // scatter pong (v final)
  __bf16* bufA = (__bf16*)take((size_t)MM * DD * 2);  // ln out / attn intermediate
  __bf16* bufB = (__bf16*)take((size_t)MM * DD * 2);  // u / q_bf16
  __bf16* bufC = (__bf16*)take((size_t)MM * DD * 2);  // kT [B,D,N]
  __bf16* bufD = (__bf16*)take((size_t)MM * DD * 2);  // v_bf16
  __bf16* kvbf = (__bf16*)take((size_t)BB * DD * DD * 2);
  float*  ksum = (float*)take((size_t)BB * DD * 4);
  float*  zrow = (float*)take((size_t)MM * 4);
  __bf16* wg_qkv = (__bf16*)take((size_t)LL * DD * D3 * 2);
  __bf16* wg_out = (__bf16*)take((size_t)LL * DD * DD * 2);
  __bf16* wa_qkv = (__bf16*)take((size_t)LL * DD * D3 * 2);
  __bf16* wa_out = (__bf16*)take((size_t)LL * DD * DD * 2);

  const float invk = 1.0f / ((float)KNN + 1e-6f);

  // -- one-time weight conversion to bf16 (idempotent, graph-safe) --
  {
    long long nq = (long long)LL * DD * D3, no = (long long)LL * DD * DD;
    f2bf_kernel<<<(nq + 255) / 256, 256, 0, stream>>>(grf_qkv,  wg_qkv, nq);
    f2bf_kernel<<<(no + 255) / 256, 256, 0, stream>>>(grf_outw, wg_out, no);
    f2bf_kernel<<<(nq + 255) / 256, 256, 0, stream>>>(attn_qkv, wa_qkv, nq);
    f2bf_kernel<<<(no + 255) / 256, 256, 0, stream>>>(attn_outw, wa_out, no);
  }

  // -- embedding --
  embed_kernel<<<(MM * DD) / 256, 256, 0, stream>>>(x, emb_w, emb_b, h);

  const size_t tbytes = (size_t)MM * DD * 4;

  for (int i = 0; i < LL; ++i) {
    const __bf16* Wg  = wg_qkv + (size_t)i * DD * D3;
    const __bf16* Wgo = wg_out + (size_t)i * DD * DD;
    const __bf16* Wa  = wa_qkv + (size_t)i * DD * D3;
    const __bf16* Wao = wa_out + (size_t)i * DD * DD;

    // ---------------- GRF layer ----------------
    ln_bf16_kernel<<<MM, 256, 0, stream>>>(h, norm_g + (2 * i) * DD,
                                           norm_b + (2 * i) * DD, bufA);
    // q -> qf (fp32), k -> t0, v later
    gemm(stream, bufA, DD, 0, Wg, D3,   0, 0, nullptr, nullptr, 0,
         qf, nullptr, nullptr, DD, 0, MM, DD, DD, 1, NN, EF_WRITE_F32);
    gemm(stream, bufA, DD, 0, Wg, D3, 256, 0, nullptr, nullptr, 0,
         t0, nullptr, nullptr, DD, 0, MM, DD, DD, 1, NN, EF_WRITE_F32);
    // k hops: t0 -> t1 -> t0 -> t1
    (void)hipMemsetAsync(t1, 0, tbytes, stream);
    scatter_mean_kernel<<<MM, 256, 0, stream>>>(t0, knn, t1, invk);
    (void)hipMemsetAsync(t0, 0, tbytes, stream);
    scatter_mean_kernel<<<MM, 256, 0, stream>>>(t1, knn, t0, invk);
    (void)hipMemsetAsync(t1, 0, tbytes, stream);
    scatter_mean_kernel<<<MM, 256, 0, stream>>>(t0, knn, t1, invk);   // k final in t1
    // v -> t0, hops: t0 -> t2 -> t0 -> t2
    gemm(stream, bufA, DD, 0, Wg, D3, 512, 0, nullptr, nullptr, 0,
         t0, nullptr, nullptr, DD, 0, MM, DD, DD, 1, NN, EF_WRITE_F32);
    (void)hipMemsetAsync(t2, 0, tbytes, stream);
    scatter_mean_kernel<<<MM, 256, 0, stream>>>(t0, knn, t2, invk);
    (void)hipMemsetAsync(t0, 0, tbytes, stream);
    scatter_mean_kernel<<<MM, 256, 0, stream>>>(t2, knn, t0, invk);
    (void)hipMemsetAsync(t2, 0, tbytes, stream);
    scatter_mean_kernel<<<MM, 256, 0, stream>>>(t0, knn, t2, invk);   // v final in t2
    // u = (q.k) * v  (bf16)
    grf_attnmul_kernel<<<MM, 256, 0, stream>>>(qf, t1, t2, bufB);
    // h += u @ out_w + out_b
    gemm(stream, bufB, DD, 0, Wgo, DD, 0, 0, grf_outb + (size_t)i * DD,
         nullptr, 0, nullptr, nullptr, h, DD, 0, MM, DD, DD, 1, NN,
         EF_BIAS | EF_RESID);

    // ---------------- linear attention layer ----------------
    ln_bf16_kernel<<<MM, 256, 0, stream>>>(h, norm_g + (2 * i + 1) * DD,
                                           norm_b + (2 * i + 1) * DD, bufA);
    // q = relu(.)+eps -> bufB (bf16)
    gemm(stream, bufA, DD, 0, Wa, D3,   0, 0, nullptr, nullptr, 0,
         nullptr, bufB, nullptr, DD, 0, MM, DD, DD, 1, NN,
         EF_RELU | EF_WRITE_BF16);
    // k = relu(.)+eps -> bufC stored transposed [B, D, N]
    gemm(stream, bufA, DD, 0, Wa, D3, 256, 0, nullptr, nullptr, 0,
         nullptr, bufC, nullptr, DD, 0, MM, DD, DD, 1, NN,
         EF_RELU | EF_TRANS_BF16);
    // v -> bufD (bf16)
    gemm(stream, bufA, DD, 0, Wa, D3, 512, 0, nullptr, nullptr, 0,
         nullptr, bufD, nullptr, DD, 0, MM, DD, DD, 1, NN,
         EF_WRITE_BF16);
    // ksum[b,d], z[b,n]
    ksum_kernel<<<BB * DD, 256, 0, stream>>>(bufC, ksum);
    z_kernel<<<MM, 256, 0, stream>>>(bufB, ksum, zrow);
    // kv[b] = kT[b] @ v[b]   (batched, K = N = 8192)
    gemm(stream, bufC, NN, (long long)DD * NN, bufD, DD, 0, (long long)NN * DD,
         nullptr, nullptr, 0, nullptr, kvbf, nullptr, DD, (long long)DD * DD,
         DD, NN, DD, BB, NN, EF_WRITE_BF16);
    // attn[b] = (q[b] @ kv[b]) * z  -> bufA (bf16), batched
    gemm(stream, bufB, DD, (long long)NN * DD, kvbf, DD, 0, (long long)DD * DD,
         nullptr, zrow, NN, nullptr, bufA, nullptr, DD, (long long)NN * DD,
         NN, DD, DD, BB, NN, EF_ROWSCALE | EF_WRITE_BF16);
    // h += attn @ out_w + out_b
    gemm(stream, bufA, DD, 0, Wao, DD, 0, 0, attn_outb + (size_t)i * DD,
         nullptr, 0, nullptr, nullptr, h, DD, 0, MM, DD, DD, 1, NN,
         EF_BIAS | EF_RESID);
  }

  // -- head --
  head_kernel<<<MM, 256, 0, stream>>>(h, head_w, head_b, out);
}
